// EfficientMaskedAttention_75557064671528
// MI455X (gfx1250) — matrix operand order repaired and checked by reading, not compiled, not run on hardware
//
#include <hip/hip_runtime.h>

// ---------------------------------------------------------------------------
// EfficientMaskedAttention for MI455X (gfx1250) — bf16 WMMA pipeline v3.1.
// b128-only fragments, async-LDS double-buffered GEMM, TDM K/V staging,
// precomputed log-bias (scale folded into Wq).
// ---------------------------------------------------------------------------

typedef __attribute__((ext_vector_type(16))) __bf16 bf16x16;
typedef __attribute__((ext_vector_type(8)))  float  f32x8;
typedef int v4i_vs __attribute__((vector_size(16)));  // gcc-style, matches builtin sig

struct U128 { unsigned x, y, z, w; };

union FragU {
  bf16x16 v;
  unsigned u32[8];
  unsigned short u16[16];
  U128 q[2];
};

union Pack8 { U128 q; unsigned u32[4]; unsigned short h[8]; };

__device__ __forceinline__ unsigned short f32_to_bf16(float f) {
  union { float f; unsigned u; } x; x.f = f;
  unsigned lsb = (x.u >> 16) & 1u;
  x.u += 0x7FFFu + lsb;  // round-to-nearest-even
  return (unsigned short)(x.u >> 16);
}

#if __has_builtin(__builtin_amdgcn_cvt_pk_bf16_f32)
__device__ __forceinline__ unsigned pack_bf16x2(float a, float b) {
  auto r = __builtin_amdgcn_cvt_pk_bf16_f32(a, b);
  unsigned u;
  __builtin_memcpy(&u, &r, 4);
  return u;
}
#else
__device__ __forceinline__ unsigned pack_bf16x2(float a, float b) {
  return (unsigned)f32_to_bf16(a) | ((unsigned)f32_to_bf16(b) << 16);
}
#endif

__device__ __forceinline__ f32x8 zero8() {
  f32x8 z;
#pragma unroll
  for (int i = 0; i < 8; ++i) z[i] = 0.0f;
  return z;
}

// Problem constants
static constexpr int kB    = 2;
static constexpr int kDim  = 512;
static constexpr int kHd   = 64;
static constexpr int kH    = 128;
static constexpr int kW    = 128;
static constexpr int kN    = 16384;  // H*W
static constexpr int kR    = 8;
static constexpr int kL    = 256;    // (H/R)*(W/R)
static constexpr int kKim  = kDim * kR * kR;  // 32768

// ---------------------------------------------------------------------------
// Async global->LDS staging (CDNA5); fallback = plain b128 copy.
// ---------------------------------------------------------------------------
#if __has_builtin(__builtin_amdgcn_global_load_async_to_lds_b128)
#define USE_ASYNC_LDS 1
#endif

__device__ __forceinline__ void stage_b128(const unsigned short* g, unsigned short* l) {
#ifdef USE_ASYNC_LDS
  __builtin_amdgcn_global_load_async_to_lds_b128(
      (__attribute__((address_space(1))) v4i_vs*)(unsigned long long)g,
      (__attribute__((address_space(3))) v4i_vs*)(unsigned)(unsigned long long)l,
      0, 0);
#else
  *reinterpret_cast<U128*>(l) = *reinterpret_cast<const U128*>(g);
#endif
}

__device__ __forceinline__ void wait_stage() {
#ifdef USE_ASYNC_LDS
#if __has_builtin(__builtin_amdgcn_s_wait_asynccnt)
  __builtin_amdgcn_s_wait_asynccnt(0);
#else
  asm volatile("s_wait_asynccnt 0x0" ::: "memory");
#endif
#endif
}

// ---------------------------------------------------------------------------
// Tensor Data Mover staging (2-D tile, bf16) — 6-arg builtin on this toolchain:
//   (uint32x4 g0, int32x8 g1, int32x4 g2, int32x4 g3, int32x8, i32 cpol)
// D# layout per CDNA5 ISA §8.3/8.4. Fallback = async-LDS loop.
// ---------------------------------------------------------------------------
#if __has_builtin(__builtin_amdgcn_tensor_load_to_lds) && \
    __has_builtin(__builtin_amdgcn_s_wait_tensorcnt)
#define USE_TDM 1
#endif

#ifdef USE_TDM
typedef unsigned tdm_v4u __attribute__((vector_size(16)));
typedef int      tdm_v8i __attribute__((vector_size(32)));
typedef int      tdm_v4i __attribute__((vector_size(16)));

// Load a [tile_d1 rows x tile_d0 elems] bf16 tile from a row-major tensor
// (row stride = stride0 elems) into LDS at lds_off, rows packed contiguously.
__device__ __forceinline__ void tdm_load_2d(
    const void* gptr, unsigned lds_off,
    unsigned tensor_d0, unsigned tensor_d1,
    unsigned tile_d0, unsigned tile_d1, unsigned stride0) {
  unsigned long long ga = (unsigned long long)gptr;
  tdm_v4u g0;
  g0[0] = 1u;                                  // count=1 (user descriptor)
  g0[1] = lds_off;                             // lds_addr (bytes)
  g0[2] = (unsigned)ga;                        // global_addr[31:0]
  g0[3] = (unsigned)(ga >> 32) | (2u << 30);   // global_addr[56:32] | type=2
  tdm_v8i g1;
  g1[0] = (int)(1u << 16);                     // wg_mask=0, data_size=1 (2B)
  g1[1] = (int)((tensor_d0 & 0xFFFFu) << 16);  // tensor_dim0[15:0] @ bits[63:48]
  g1[2] = (int)((tensor_d0 >> 16) | ((tensor_d1 & 0xFFFFu) << 16));
  g1[3] = (int)((tensor_d1 >> 16) | (tile_d0 << 16));   // tile_dim0 @ [127:112]
  g1[4] = (int)(tile_d1 & 0xFFFFu);            // tile_dim1; tile_dim2 = 0
  g1[5] = (int)stride0;                        // tensor_dim0_stride[31:0]
  g1[6] = 0;                                   // stride0[47:32], stride1[15:0]
  g1[7] = 0;
  tdm_v4i z4 = {0, 0, 0, 0};                   // groups 2/3 unused (<=2-D)
  tdm_v8i z8 = {0, 0, 0, 0, 0, 0, 0, 0};
  __builtin_amdgcn_tensor_load_to_lds(g0, g1, z4, z4, z8, 0);
}
#endif

// ---------------------------------------------------------------------------
// WMMA fragment loaders (ISA 7.12.2, wave32). Both are 2x b128 loads.
//   A 16x32: tile stored [row][K] row-major, ldk elems.
//   B 32x16: tile stored TRANSPOSED [col][K] row-major, ldk elems.
// ---------------------------------------------------------------------------
__device__ __forceinline__ bf16x16 load_a_frag(const unsigned short* base, int ldk) {
  int lane = threadIdx.x & 31;
  int lo = lane & 15, hi = lane >> 4;
  const unsigned short* p = base + lo * ldk + hi * 8;
  FragU u;
  u.q[0] = *reinterpret_cast<const U128*>(p);
  u.q[1] = *reinterpret_cast<const U128*>(p + 16);
  return u.v;
}

__device__ __forceinline__ bf16x16 load_b_frag(const unsigned short* base, int ldk) {
  int lane = threadIdx.x & 31;
  int lo = lane & 15, hi = lane >> 4;
  const unsigned short* p = base + lo * ldk + hi * 8;
  FragU u;
  u.q[0] = *reinterpret_cast<const U128*>(p);
  u.q[1] = *reinterpret_cast<const U128*>(p + 16);
  return u.v;
}

// ---------------------------------------------------------------------------
// f32 -> bf16 elementwise with scale (scale folds attention 1/sqrt(hd) into Wq)
// ---------------------------------------------------------------------------
__global__ __launch_bounds__(256) void cvt_bf16_kernel(
    const float* __restrict__ src, unsigned short* __restrict__ dst, int n,
    float scale) {
  int i = blockIdx.x * 256 + threadIdx.x;
  if (i < n) dst[i] = f32_to_bf16(src[i] * scale);
}

// ---------------------------------------------------------------------------
// LDS-tiled transpose + convert: x [b][C][N] f32 -> Xt [b][N][C] bf16
// ---------------------------------------------------------------------------
__global__ __launch_bounds__(256) void transpose_cvt_kernel(
    const float* __restrict__ src, unsigned short* __restrict__ dst,
    int C, int N) {
  __shared__ unsigned short t[64][65];
  const int n0 = blockIdx.x * 64, c0 = blockIdx.y * 64, b = blockIdx.z;
  const float* s = src + (long long)b * C * N;
  unsigned short* d = dst + (long long)b * N * C;
  const int a = threadIdx.x & 63, r = threadIdx.x >> 6;
#pragma unroll
  for (int j = 0; j < 16; ++j) {
    int c = r + j * 4;
    t[c][a] = f32_to_bf16(s[(long long)(c0 + c) * N + n0 + a]);
  }
  __syncthreads();
#pragma unroll
  for (int j = 0; j < 16; ++j) {
    int n = r + j * 4;
    d[(long long)(n0 + n) * C + c0 + a] = t[a][n];
  }
}

// ---------------------------------------------------------------------------
// LDS-tiled bf16 transpose: src [b][C][L] -> dst [b][L][C]
// ---------------------------------------------------------------------------
__global__ __launch_bounds__(256) void transpose_bf16_kernel(
    const unsigned short* __restrict__ src, unsigned short* __restrict__ dst,
    int C, int L, long long srcStride, long long dstStride) {
  __shared__ unsigned short t[64][65];
  const int l0 = blockIdx.x * 64, c0 = blockIdx.y * 64, b = blockIdx.z;
  const unsigned short* s = src + (long long)b * srcStride;
  unsigned short* d = dst + (long long)b * dstStride;
  const int a = threadIdx.x & 63, r = threadIdx.x >> 6;
#pragma unroll
  for (int j = 0; j < 16; ++j) {
    int c = r + j * 4;
    t[c][a] = s[(long long)(c0 + c) * L + l0 + a];
  }
  __syncthreads();
#pragma unroll
  for (int j = 0; j < 16; ++j) {
    int l = r + j * 4;
    d[(long long)(l0 + l) * C + c0 + a] = t[a][l];
  }
}

// ---------------------------------------------------------------------------
// im2col, emitted TRANSPOSED: XcolT[b][l][c*64+dy*8+dx]  (K contiguous)
// ---------------------------------------------------------------------------
__global__ __launch_bounds__(256) void im2col_t_kernel(
    const float* __restrict__ x, unsigned short* __restrict__ xcol, int total) {
  int i = blockIdx.x * 256 + threadIdx.x;
  if (i >= total) return;
  int k = i & (kKim - 1);
  int l = (i >> 15) & (kL - 1);
  int b = i >> 23;
  int c = k >> 6, dy = (k >> 3) & 7, dx = k & 7;
  int hr = l >> 4, wr = l & 15;
  xcol[i] = f32_to_bf16(
      x[(((long long)b * kDim + c) * kH + hr * kR + dy) * kW + wr * kR + dx]);
}

// ---------------------------------------------------------------------------
// 8x8 average pool of the mask -> mask_down[b][l] (f32)
// ---------------------------------------------------------------------------
__global__ __launch_bounds__(256) void mask_down_kernel(
    const float* __restrict__ mask, float* __restrict__ md) {
  int i = blockIdx.x * 256 + threadIdx.x;
  if (i >= kB * kL) return;
  int b = i >> 8, l = i & (kL - 1);
  int hr = l >> 4, wr = l & 15;
  float s = 0.0f;
#pragma unroll
  for (int dy = 0; dy < kR; ++dy)
#pragma unroll
    for (int dx = 0; dx < kR; ++dx)
      s += mask[(long long)b * kN + (hr * kR + dy) * kW + (wr * kR + dx)];
  md[i] = s * (1.0f / 64.0f);
}

// ---------------------------------------------------------------------------
// Precomputed scaled log-bias: biasS[b][n][l] = scale*log(m[n]*md[l]+eps)
// (bias batch reuse is 8x across heads — hoist the transcendental work)
// ---------------------------------------------------------------------------
__global__ __launch_bounds__(256) void bias_kernel(
    const float* __restrict__ mask, const float* __restrict__ md,
    float* __restrict__ biasS) {
  int i = blockIdx.x * 256 + threadIdx.x;  // 2 * 2^22 total
  int l = i & (kL - 1);
  int n = (i >> 8) & (kN - 1);
  int b = i >> 22;
  if (b >= kB) return;
  biasS[i] = 0.125f * __logf(mask[(long long)b * kN + n] * md[b * kL + l] + 1e-8f);
}

// ---------------------------------------------------------------------------
// GEMM: C[bz] = A(MxK) x B[bz], B stored COLUMN-MAJOR (Bt[N][K]).
// 256 thr = 8 waves, tile 128x128, wave tile 32x64; double-buffered async LDS.
// OUT_MODE: 0 = f32 [M][N], 1 = bf16 [M][N], 2 = bf16 TRANSPOSED [N][M]
// ---------------------------------------------------------------------------
template <int OUT_MODE>
__global__ __launch_bounds__(256) void gemm_bt_kernel(
    const unsigned short* __restrict__ A, const unsigned short* __restrict__ Bt,
    void* __restrict__ C, int M, int N, int K,
    long long strideB, long long strideC) {
  __shared__ unsigned short As[2][128 * 32];  // [row][k]
  __shared__ unsigned short Bs[2][128 * 32];  // [col][k]

  const int tid  = threadIdx.x;
  const int lane = tid & 31;
  const int wave = tid >> 5;
  const int lo   = lane & 15, hi = lane >> 4;
  const int n0   = blockIdx.x * 128;
  const int m0   = blockIdx.y * 128;
  const int bz   = blockIdx.z;
  const unsigned short* Bb = Bt + (long long)bz * strideB;

  const int wm = wave >> 1;  // 0..3
  const int wn = wave & 1;   // 0..1

  f32x8 acc[2][4];
#pragma unroll
  for (int mi = 0; mi < 2; ++mi)
#pragma unroll
    for (int ni = 0; ni < 4; ++ni) acc[mi][ni] = zero8();

  const int row = (tid >> 2);      // 0..63
  const int kc  = (tid & 3) << 3;  // 0,8,16,24

  auto stage_chunk = [&](int ck, int buf) {
    int kk = ck * 32;
#pragma unroll
    for (int r2 = 0; r2 < 2; ++r2) {
      int rr = row + r2 * 64;
      stage_b128(&A[(long long)(m0 + rr) * K + kk + kc], &As[buf][rr * 32 + kc]);
      stage_b128(&Bb[(long long)(n0 + rr) * K + kk + kc], &Bs[buf][rr * 32 + kc]);
    }
  };

  const int nchunks = K >> 5;
  stage_chunk(0, 0);
  wait_stage();
  __syncthreads();

  for (int ck = 0; ck < nchunks; ++ck) {
    const int cur = ck & 1;
    if (ck + 1 < nchunks) stage_chunk(ck + 1, cur ^ 1);  // overlap copy/compute

    bf16x16 af[2], bfv[4];
#pragma unroll
    for (int mi = 0; mi < 2; ++mi)
      af[mi] = load_a_frag(&As[cur][(wm * 32 + mi * 16) * 32], 32);
#pragma unroll
    for (int ni = 0; ni < 4; ++ni)
      bfv[ni] = load_b_frag(&Bs[cur][(wn * 64 + ni * 16) * 32], 32);

#pragma unroll
    for (int mi = 0; mi < 2; ++mi)
#pragma unroll
      for (int ni = 0; ni < 4; ++ni)
        acc[mi][ni] = __builtin_amdgcn_wmma_f32_16x16x32_bf16(
            false, af[mi], false, bfv[ni], (short)0, acc[mi][ni], false, false);

    wait_stage();
    __syncthreads();
  }

#pragma unroll
  for (int mi = 0; mi < 2; ++mi)
#pragma unroll
    for (int ni = 0; ni < 4; ++ni) {
      if (OUT_MODE == 2) {
        int coln  = n0 + wn * 64 + ni * 16 + lo;
        int mbase = m0 + wm * 32 + mi * 16 + hi * 8;
        Pack8 pk;
#pragma unroll
        for (int r = 0; r < 4; ++r)
          pk.u32[r] = pack_bf16x2(acc[mi][ni][2 * r], acc[mi][ni][2 * r + 1]);
        *reinterpret_cast<U128*>(&reinterpret_cast<unsigned short*>(C)[
            (long long)bz * strideC + (long long)coln * M + mbase]) = pk.q;
      } else {
#pragma unroll
        for (int r = 0; r < 8; ++r) {
          int mrow = m0 + wm * 32 + mi * 16 + r + hi * 8;
          int ncol = n0 + wn * 64 + ni * 16 + lo;
          long long idx = (long long)bz * strideC + (long long)mrow * N + ncol;
          if (OUT_MODE == 1)
            reinterpret_cast<unsigned short*>(C)[idx] = f32_to_bf16(acc[mi][ni][r]);
          else
            reinterpret_cast<float*>(C)[idx] = acc[mi][ni][r];
        }
      }
    }
}

// ---------------------------------------------------------------------------
// Fused attention per (bh, 64-row block of N):
//   S = Qs K^T + biasS; softmax over L=256; O = P V.  (scale pre-folded)
//   Qt : bf16 [b][N][512], Kt : bf16 [b][L][512], KV : bf16 [b][1024][L]
//   AO : bf16 [b][N][512]  ([n][c] = Bt layout for the Wo GEMM)
// Bias batch index = bh % B (faithful to reference's tile()).
// ---------------------------------------------------------------------------
__global__ __launch_bounds__(128) void attn_kernel(
    const unsigned short* __restrict__ Qt, const unsigned short* __restrict__ Kt,
    const unsigned short* __restrict__ KV, const float* __restrict__ biasS,
    unsigned short* __restrict__ AO) {
  __shared__ unsigned short kv_lds[kHd * kL];   // 32 KB: Kt[l][64], then V[c][l]
  __shared__ unsigned short p_lds[4][16 * kL];  // 32 KB: per-wave P [n][l]

  const int tid  = threadIdx.x;
  const int lane = tid & 31;
  const int wave = tid >> 5;
  const int lo   = lane & 15, hi = lane >> 4;
  const int bh = blockIdx.y;
  const int b  = bh >> 3;        // heads = 8
  const int h  = bh & 7;
  const int bb = bh & (kB - 1);  // bias batch = bh % B
  const int n0 = blockIdx.x * 64;
  const int nw = n0 + wave * 16;

  // ---- stage K^T tile for this head: kv_lds[l][64] ----
#ifdef USE_TDM
  if (wave == 0) {
    tdm_load_2d(Kt + (long long)b * kL * kDim + h * kHd,
                (unsigned)(unsigned long long)&kv_lds[0],
                kDim, kL, kHd, kL, kDim);
    __builtin_amdgcn_s_wait_tensorcnt(0);
  }
#else
  {
    const unsigned short* src = Kt + (long long)b * kL * kDim + h * kHd;
    for (int i = tid; i < kL * 8; i += 128) {
      int r = i >> 3, w = i & 7;
      stage_b128(src + (long long)r * kDim + w * 8, &kv_lds[r * 64 + w * 8]);
    }
    wait_stage();
  }
#endif
  __syncthreads();

  // ---- Q A-fragments from global (2x b128 each) ----
  const unsigned short* qb = Qt + ((long long)b * kN + nw) * kDim + h * kHd;
  bf16x16 qf[2];
#pragma unroll
  for (int kc = 0; kc < 2; ++kc) qf[kc] = load_a_frag(qb + kc * 32, kDim);

  // ---- S = Q K^T : 16 l-tiles x 2 k-chunks = 32 WMMAs ----
  f32x8 s[16];
#pragma unroll
  for (int t = 0; t < 16; ++t) s[t] = zero8();
#pragma unroll
  for (int t = 0; t < 16; ++t)
#pragma unroll
    for (int kc = 0; kc < 2; ++kc) {
      bf16x16 kf = load_b_frag(&kv_lds[(t * 16) * 64 + kc * 32], 64);
      s[t] = __builtin_amdgcn_wmma_f32_16x16x32_bf16(
          false, qf[kc], false, kf, (short)0, s[t], false, false);
    }

  // ---- add precomputed scaled bias (C layout: row n = r+8hi, col l = t*16+lo)
  const float* bsrow = biasS + ((long long)bb * kN + nw + hi * 8) * kL + lo;
#pragma unroll
  for (int t = 0; t < 16; ++t)
#pragma unroll
    for (int r = 0; r < 8; ++r)
      s[t][r] += bsrow[(long long)r * kL + t * 16];

  // ---- softmax over L = 256 (registers + 16-lane shfl reductions) ----
#pragma unroll
  for (int r = 0; r < 8; ++r) {
    float m = -3.0e38f;
#pragma unroll
    for (int t = 0; t < 16; ++t) m = fmaxf(m, s[t][r]);
#pragma unroll
    for (int off = 8; off >= 1; off >>= 1) m = fmaxf(m, __shfl_xor(m, off, 32));
    float acc = 0.0f;
#pragma unroll
    for (int t = 0; t < 16; ++t) {
      float p = __expf(s[t][r] - m);
      s[t][r] = p;
      acc += p;
    }
#pragma unroll
    for (int off = 8; off >= 1; off >>= 1) acc += __shfl_xor(acc, off, 32);
    float rinv = 1.0f / acc;
#pragma unroll
    for (int t = 0; t < 16; ++t)
      p_lds[wave][(r + hi * 8) * kL + t * 16 + lo] = f32_to_bf16(s[t][r] * rinv);
  }

  __syncthreads();  // all P written, all K reads done

  // ---- stage V (contiguous [c][l] block) over K's LDS ----
#ifdef USE_TDM
  if (wave == 0) {
    tdm_load_2d(KV + ((long long)b * 2 * kDim + kDim + h * kHd) * kL,
                (unsigned)(unsigned long long)&kv_lds[0],
                kL, kHd, kL, kHd, kL);
    __builtin_amdgcn_s_wait_tensorcnt(0);
  }
#else
  {
    const unsigned short* vsrc =
        KV + ((long long)b * 2 * kDim + kDim + h * kHd) * kL;
    for (int i = tid; i < (kHd * kL) / 8; i += 128)
      stage_b128(vsrc + i * 8, &kv_lds[i * 8]);
    wait_stage();
  }
#endif
  __syncthreads();

  // ---- O = P V : 4 c-tiles x 8 l-chunks = 32 WMMAs ----
  f32x8 o[4];
#pragma unroll
  for (int ct = 0; ct < 4; ++ct) o[ct] = zero8();
#pragma unroll
  for (int lc = 0; lc < 8; ++lc) {
    bf16x16 pa = load_a_frag(&p_lds[wave][lc * 32], kL);
#pragma unroll
    for (int ct = 0; ct < 4; ++ct) {
      bf16x16 vb = load_b_frag(&kv_lds[(ct * 16) * kL + lc * 32], kL);
      o[ct] = __builtin_amdgcn_wmma_f32_16x16x32_bf16(
          false, pa, false, vb, (short)0, o[ct], false, false);
    }
  }

  // ---- store O as [n][c] (Bt layout for the Wo GEMM) ----
  unsigned short* ob = AO + ((long long)b * kN + nw) * kDim + h * kHd;
#pragma unroll
  for (int ct = 0; ct < 4; ++ct)
#pragma unroll
    for (int r = 0; r < 8; ++r)
      ob[(long long)(r + hi * 8) * kDim + ct * 16 + lo] = f32_to_bf16(o[ct][r]);
}

// ---------------------------------------------------------------------------
// Host launch
// ---------------------------------------------------------------------------
extern "C" void kernel_launch(void* const* d_in, const int* in_sizes, int n_in,
                              void* d_out, int out_size, void* d_ws, size_t ws_size,
                              hipStream_t stream) {
  const float* x    = (const float*)d_in[0];  // [2,512,128,128]
  const float* mask = (const float*)d_in[1];  // [2,1,128,128]
  const float* Wq   = (const float*)d_in[2];  // [512,512,1,1]
  const float* Wkv  = (const float*)d_in[3];  // [1024,512,8,8]
  const float* Wo   = (const float*)d_in[4];  // [512,512,1,1]
  float* out = (float*)d_out;                 // [2,512,128,128]

  char* ws = (char*)d_ws;
  auto take = [&](size_t bytes) {
    char* p = ws;
    ws += (bytes + 255) & ~size_t(255);
    return p;
  };
  const long long nX   = (long long)kB * kDim * kN;
  const long long nWq  = (long long)kDim * kDim;
  const long long nWkv = (long long)2 * kDim * kKim;
  const long long nCol = (long long)kB * kL * kKim;
  const long long nKV  = (long long)kB * 2 * kDim * kL;
  const long long nKt  = (long long)kB * kL * kDim;
  const long long nBs  = (long long)kB * kN * kL;

  unsigned short* Xt     = (unsigned short*)take(nX * 2);    // [b][n][c]
  unsigned short* Wq_bf  = (unsigned short*)take(nWq * 2);
  unsigned short* Wkv_bf = (unsigned short*)take(nWkv * 2);
  unsigned short* Wo_bf  = (unsigned short*)take(nWq * 2);
  unsigned short* XcolT  = (unsigned short*)take(nCol * 2);  // [b][l][k]
  float*          md     = (float*)take(kB * kL * 4);
  float*          biasS  = (float*)take(nBs * 4);            // [b][n][l]
  unsigned short* Qt     = (unsigned short*)take(nX * 2);    // [b][n][c]
  unsigned short* KV_bf  = (unsigned short*)take(nKV * 2);   // [b][1024][l]
  unsigned short* Kt     = (unsigned short*)take(nKt * 2);   // [b][l][c]
  unsigned short* AO     = (unsigned short*)take(nX * 2);    // [b][n][c]

  // 1) conversions / layout transforms
  {
    dim3 g(kN / 64, kDim / 64, kB);
    transpose_cvt_kernel<<<g, 256, 0, stream>>>(x, Xt, kDim, kN);
  }
  cvt_bf16_kernel<<<(int)((nWq + 255) / 256), 256, 0, stream>>>(Wq, Wq_bf, (int)nWq, 0.125f);
  cvt_bf16_kernel<<<(int)((nWkv + 255) / 256), 256, 0, stream>>>(Wkv, Wkv_bf, (int)nWkv, 1.0f);
  cvt_bf16_kernel<<<(int)((nWq + 255) / 256), 256, 0, stream>>>(Wo, Wo_bf, (int)nWq, 1.0f);
  im2col_t_kernel<<<(int)((nCol + 255) / 256), 256, 0, stream>>>(x, XcolT, (int)nCol);
  mask_down_kernel<<<2, 256, 0, stream>>>(mask, md);
  bias_kernel<<<(int)((nBs + 255) / 256), 256, 0, stream>>>(mask, md, biasS);

  // 2) Qt[b][n][512] = (scale * Wq x X)^T
  {
    dim3 g(kN / 128, kDim / 128, kB);
    gemm_bt_kernel<2><<<g, 256, 0, stream>>>(
        Wq_bf, Xt, Qt, kDim, kN, kDim,
        (long long)kN * kDim, (long long)kN * kDim);
  }
  // 3) KV[b][1024][256] = Wkv x Xcol
  {
    dim3 g(kL / 128, (2 * kDim) / 128, kB);
    gemm_bt_kernel<1><<<g, 256, 0, stream>>>(
        Wkv_bf, XcolT, KV_bf, 2 * kDim, kL, kKim,
        (long long)kL * kKim, (long long)2 * kDim * kL);
  }
  // 4) Kt[b][l][512] = K^T (first 512 channels of KV)
  {
    dim3 g(kL / 64, kDim / 64, kB);
    transpose_bf16_kernel<<<g, 256, 0, stream>>>(
        KV_bf, Kt, kDim, kL, (long long)2 * kDim * kL, (long long)kL * kDim);
  }
  // 5) fused masked attention -> AO[b][n][512]
  {
    dim3 g(kN / 64, kB * 8);
    attn_kernel<<<g, 128, 0, stream>>>(Qt, Kt, KV_bf, biasS, AO);
  }
  // 6) out[b][512][N] = Wo x AO^T  (f32 straight to d_out)
  {
    dim3 g(kN / 128, kDim / 128, kB);
    gemm_bt_kernel<0><<<g, 256, 0, stream>>>(
        Wo_bf, AO, out, kDim, kN, kDim,
        (long long)kN * kDim, (long long)kN * kDim);
  }
  (void)in_sizes; (void)n_in; (void)out_size; (void)ws_size;
}
